// XlMultiheadAttention_21706764714682
// MI455X (gfx1250) — matrix operand order repaired and checked by reading, not compiled, not run on hardware
//
#include <hip/hip_runtime.h>
#include <hip/hip_bf16.h>

// Transformer-XL attention for gfx1250 (MI455X), wave32 + WMMA bf16.
//
// Exact simplification from the reference's _rel_shift: the shifted term_bd
// broadcasts along the softmax axis, so it cancels inside softmax. Only the
// q/k/v projections, softmax((q+rel_u)K^T/8), att@V, out-proj and the
// head-mean attention weights are live. rel_v / rel_proj_w / sin_pos_enc are
// dead inputs.
//
// Data flow (bf16 operand caches built once, consumed by WMMA many times):
//   gemm<BF16> : qb  = bf16(query@Wq^T + bq + rel_u_flat)   [l*NB+n][e]
//   gemm<BF16> : kb  = bf16(key  @Wk^T + bk)                [s*NB+n][e]
//   gemm<BF16T>: vT  = bf16(value@Wv^T + bv) transposed     [n][h][d][s]
//   attn_stats : online softmax row (max, sumexp)
//   attn_pass2 : probs -> att@V (WMMA) + head-mean att_weights (LDS reduce)
//   gemm<F32>  : out = outpre@Wo^T + bo

typedef __attribute__((ext_vector_type(16))) __bf16 bf16x16;
typedef __attribute__((ext_vector_type(8)))  __bf16 bf16x8;
typedef __attribute__((ext_vector_type(8)))  float  f32x8;

namespace {

constexpr int L  = 2048;   // query length
constexpr int S  = 2048;   // key length
constexpr int NB = 4;      // batch
constexpr int E  = 512;    // embed dim
constexpr int H  = 8;      // heads
constexpr int Dh = 64;     // head dim
constexpr float SCALE = 0.125f;  // 1/sqrt(Dh)

union frag16 { bf16x16 v; bf16x8 h[2]; };

__device__ inline f32x8 wmma_bf16(bf16x16 a, bf16x16 b, f32x8 c) {
  // v_wmma_f32_16x16x32_bf16 : D = A(16x32) * B(32x16) + C
  return __builtin_amdgcn_wmma_f32_16x16x32_bf16(
      /*neg_a=*/false, a, /*neg_b=*/false, b,
      /*c_mod=*/(short)0, c, /*reuse_a=*/false, /*reuse_b=*/false);
}

// ---- fp32-source fragment loaders (projection GEMMs) ---------------------
// A fragment 16(M)x32(K): lane%16 = M row; lanes>=16 shift K by 8;
// VGPR pair j holds K = (j>=4?16:0) + (j&3)*2 (+lane-half offset).
__device__ inline bf16x16 ldfrag_a_f32(const float* A, int lda, int k0, int lane) {
  bf16x16 f;
  const float* p = A + (lane & 15) * lda;
  const int kb = k0 + ((lane >> 4) << 3);
#pragma unroll
  for (int j = 0; j < 8; ++j) {
    const int k = kb + ((j >> 2) << 4) + ((j & 3) << 1);
    f[2 * j]     = (__bf16)p[k];
    f[2 * j + 1] = (__bf16)p[k + 1];
  }
  return f;
}

// B fragment for C = A*B^T, B row-major [n][k]: lane%16 = N; lane half = K+8, elements 8..15 at K+16.
__device__ inline bf16x16 ldfrag_bt_f32(const float* B, int ldb, int k0, int lane) {
  bf16x16 f;
  const float* p = B + (lane & 15) * ldb + k0 + ((lane >> 4) << 3);
#pragma unroll
  for (int j = 0; j < 16; ++j) f[j] = (__bf16)p[j < 8 ? j : j + 8];
  return f;
}

// ---- bf16-source fragment loaders (attention hot loops, b128 loads) ------
__device__ inline bf16x16 ldfrag_a16(const __bf16* A, int lda, int k0, int lane) {
  const __bf16* p = A + (lane & 15) * lda + k0 + ((lane >> 4) << 3);
  frag16 f;
  f.h[0] = *(const bf16x8*)p;          // K kb .. kb+7
  f.h[1] = *(const bf16x8*)(p + 16);   // K kb+16 .. kb+23
  return f.v;
}

__device__ inline bf16x16 ldfrag_bt16(const __bf16* B, int ldb, int k0, int lane) {
  const __bf16* p = B + (lane & 15) * ldb + k0 + ((lane >> 4) << 3);
  frag16 f;
  f.h[0] = *(const bf16x8*)p;          // K kb .. kb+7
  f.h[1] = *(const bf16x8*)(p + 16);   // K kb+16 .. kb+23
  return f.v;
}

// ---- Kernel 1: C[M,Nc] = A[M,K] @ B[Nc,K]^T + bias (+bias2) --------------
// MODE 0: fp32 out, row-major [m][nc]
// MODE 1: bf16 out, row-major [m][nc]           (q / k operand caches)
// MODE 2: bf16 out, transposed [n][h][d][s]     (vT operand cache)
template <int MODE>
__global__ __launch_bounds__(256) void gemm_bt_bias(
    const float* __restrict__ A, const float* __restrict__ B,
    const float* __restrict__ bias, const float* __restrict__ bias2,
    void* __restrict__ C, int M, int Nc, int K) {
  const int lane = threadIdx.x & 31;
  const int wave = threadIdx.x >> 5;
  const int tile = blockIdx.x * (blockDim.x >> 5) + wave;
  const int ntn  = Nc >> 4;
  const int tm   = tile / ntn;
  const int tn   = tile - tm * ntn;
  if (tm * 16 >= M) return;  // wave-uniform

  const float* Ap = A + tm * 16 * K;
  const float* Bp = B + tn * 16 * K;
  f32x8 c = {};
  for (int k0 = 0; k0 < K; k0 += 32) {
    bf16x16 a = ldfrag_a_f32(Ap, K, k0, lane);
    bf16x16 b = ldfrag_bt_f32(Bp, K, k0, lane);
    c = wmma_bf16(a, b, c);
  }
  const int nc = tn * 16 + (lane & 15);
  float bv = bias[nc];
  if (bias2) bv += bias2[nc];
  const int mb = tm * 16 + ((lane >> 4) << 3);
  if constexpr (MODE == 0) {
    float* Cp = (float*)C;
#pragma unroll
    for (int r = 0; r < 8; ++r) Cp[(mb + r) * Nc + nc] = c[r] + bv;
  } else if constexpr (MODE == 1) {
    __bf16* Cp = (__bf16*)C;
#pragma unroll
    for (int r = 0; r < 8; ++r) Cp[(mb + r) * Nc + nc] = (__bf16)(c[r] + bv);
  } else {
    __bf16* Cp = (__bf16*)C;
    const int h = nc >> 6, d = nc & 63;
#pragma unroll
    for (int r = 0; r < 8; ++r) {
      const int m = mb + r;
      const int s = m >> 2, n = m & 3;  // row = s*NB + n
      Cp[((n * H + h) * Dh + d) * S + s] = (__bf16)(c[r] + bv);
    }
  }
}

// ---- Kernel 2: online softmax stats (rowmax m, sumexp) per (n,h,l) -------
// One wave per (n, h, 16-row l-tile); 32 key positions per iteration so one
// shfl_xor reduction tree serves two WMMA score tiles.
__global__ __launch_bounds__(256) void attn_stats(
    const __bf16* __restrict__ qb, const __bf16* __restrict__ kb,
    float* __restrict__ mbuf, float* __restrict__ sbuf) {
  const int lane = threadIdx.x & 31;
  const int wave = threadIdx.x >> 5;
  const int gw = blockIdx.x * (blockDim.x >> 5) + wave;   // 0..4095
  const int lt = gw & (L / 16 - 1);
  const int nh = gw >> 7;
  const int h = nh & (H - 1);
  const int n = nh >> 3;
  const int l0 = lt * 16;
  const int ld = NB * E;
  const __bf16* qp = qb + (l0 * NB + n) * E + h * Dh;
  const __bf16* kp = kb + n * E + h * Dh;

  const bf16x16 a0 = ldfrag_a16(qp, ld, 0, lane);
  const bf16x16 a1 = ldfrag_a16(qp, ld, 32, lane);

  float mr[8], ss[8];
#pragma unroll
  for (int r = 0; r < 8; ++r) { mr[r] = -3.0e38f; ss[r] = 0.f; }

  for (int s0 = 0; s0 < S; s0 += 32) {
    const __bf16* kt0 = kp + s0 * ld;
    const __bf16* kt1 = kt0 + 16 * ld;
    f32x8 c0 = {}, c1 = {};
    c0 = wmma_bf16(a0, ldfrag_bt16(kt0, ld, 0, lane), c0);
    c0 = wmma_bf16(a1, ldfrag_bt16(kt0, ld, 32, lane), c0);
    c1 = wmma_bf16(a0, ldfrag_bt16(kt1, ld, 0, lane), c1);
    c1 = wmma_bf16(a1, ldfrag_bt16(kt1, ld, 32, lane), c1);
#pragma unroll
    for (int r = 0; r < 8; ++r) {
      const float x0 = c0[r] * SCALE, x1 = c1[r] * SCALE;
      float tmax = fmaxf(x0, x1);
#pragma unroll
      for (int off = 8; off >= 1; off >>= 1)
        tmax = fmaxf(tmax, __shfl_xor(tmax, off, 32));
      const float nm = fmaxf(mr[r], tmax);
      float e = __expf(x0 - nm) + __expf(x1 - nm);
#pragma unroll
      for (int off = 8; off >= 1; off >>= 1)
        e += __shfl_xor(e, off, 32);
      ss[r] = ss[r] * __expf(mr[r] - nm) + e;
      mr[r] = nm;
    }
  }
  if ((lane & 15) == 0) {  // lanes 0 and 16 write rows 0-7 / 8-15
    const int base = (n * H + h) * L + l0 + ((lane >> 4) << 3);
#pragma unroll
    for (int r = 0; r < 8; ++r) { mbuf[base + r] = mr[r]; sbuf[base + r] = ss[r]; }
  }
}

// ---- Kernel 3: probabilities, att@V, head-mean attention weights ---------
// Block = 8 waves = 8 heads for one (n, 16-row l-tile). Probabilities staged
// in LDS: (a) summed over the 8 waves -> att_weights (deterministic, no
// atomics), (b) re-read in A-fragment layout for the P(16x32)@V(32x64) WMMAs.
__global__ __launch_bounds__(256) void attn_pass2(
    const __bf16* __restrict__ qb, const __bf16* __restrict__ kb,
    const __bf16* __restrict__ vt, const float* __restrict__ mbuf,
    const float* __restrict__ sbuf, float* __restrict__ outpre,
    float* __restrict__ attw) {
  __shared__ float pshare[H][16][32];  // 16 KB
  const int lane = threadIdx.x & 31;
  const int h = threadIdx.x >> 5;      // wave index == head
  const int l0 = blockIdx.x * 16;
  const int n = blockIdx.y;
  const int ld = NB * E;
  const __bf16* qp = qb + (l0 * NB + n) * E + h * Dh;
  const __bf16* kp = kb + n * E + h * Dh;
  const __bf16* vp = vt + (n * H + h) * Dh * S;   // [d][s], ld = S

  const bf16x16 a0 = ldfrag_a16(qp, ld, 0, lane);
  const bf16x16 a1 = ldfrag_a16(qp, ld, 32, lane);

  const int rr = (lane >> 4) << 3;     // local row base for this lane half
  const int sbase = (n * H + h) * L + l0 + rr;
  float mr[8], is[8];
#pragma unroll
  for (int r = 0; r < 8; ++r) { mr[r] = mbuf[sbase + r]; is[r] = 1.0f / sbuf[sbase + r]; }

  const f32x8 zero = {};
  f32x8 o[4];
#pragma unroll
  for (int d = 0; d < 4; ++d) o[d] = zero;

  for (int s0 = 0; s0 < S; s0 += 32) {
#pragma unroll
    for (int t = 0; t < 2; ++t) {
      const __bf16* kt = kp + (s0 + t * 16) * ld;
      f32x8 c = {};
      c = wmma_bf16(a0, ldfrag_bt16(kt, ld, 0, lane), c);
      c = wmma_bf16(a1, ldfrag_bt16(kt, ld, 32, lane), c);
      const int col = t * 16 + (lane & 15);
#pragma unroll
      for (int r = 0; r < 8; ++r)
        pshare[h][rr + r][col] = __expf(c[r] * SCALE - mr[r]) * is[r];
    }
    __syncthreads();

    // att_weights[n, l, s] = mean over heads (LDS reduction across waves)
    for (int e2 = threadIdx.x; e2 < 16 * 32; e2 += 256) {
      const int row = e2 >> 5, col = e2 & 31;
      float acc = 0.f;
#pragma unroll
      for (int w = 0; w < H; ++w) acc += pshare[w][row][col];
      attw[(n * L + l0 + row) * S + s0 + col] = acc * (1.0f / H);
    }

    // Re-read own prob tile in A-fragment layout, then O += P @ V
    bf16x16 pa;
    {
      const float* pr = &pshare[h][lane & 15][0];
      const int kb2 = (lane >> 4) << 3;
#pragma unroll
      for (int j = 0; j < 8; ++j) {
        const int k = kb2 + ((j >> 2) << 4) + ((j & 3) << 1);
        pa[2 * j]     = (__bf16)pr[k];
        pa[2 * j + 1] = (__bf16)pr[k + 1];
      }
    }
#pragma unroll
    for (int dt = 0; dt < 4; ++dt) {
      bf16x16 bv = ldfrag_bt16(vp + (dt * 16) * S + s0, S, 0, lane);
      o[dt] = wmma_bf16(pa, bv, o[dt]);
    }
    __syncthreads();
  }

  // Write attention output in [l, n, e] layout for the final projection.
  const int dc = lane & 15;
#pragma unroll
  for (int dt = 0; dt < 4; ++dt)
#pragma unroll
    for (int r = 0; r < 8; ++r)
      outpre[((l0 + rr + r) * NB + n) * E + h * Dh + dt * 16 + dc] = o[dt][r];
}

}  // namespace

extern "C" void kernel_launch(void* const* d_in, const int* in_sizes, int n_in,
                              void* d_out, int out_size, void* d_ws, size_t ws_size,
                              hipStream_t stream) {
  (void)in_sizes; (void)n_in; (void)out_size; (void)ws_size;
  const float* query = (const float*)d_in[0];
  const float* key   = (const float*)d_in[1];
  const float* value = (const float*)d_in[2];
  const float* in_w  = (const float*)d_in[4];   // (3E, E)
  const float* in_b  = (const float*)d_in[5];   // (3E,)
  const float* out_w = (const float*)d_in[6];   // (E, E)
  const float* out_b = (const float*)d_in[7];   // (E,)
  const float* rel_u = (const float*)d_in[9];   // (H, D) flat == length-E bias
  // d_in[3] sin_pos_enc, d_in[8] rel_proj_w, d_in[10] rel_v: dead (see header)

  float* out  = (float*)d_out;                  // (L, N, E)
  float* attw = out + (size_t)L * NB * E;       // (N, L, S)

  constexpr size_t QN = (size_t)L * NB * E;     // 4194304 elements
  __bf16* qb = (__bf16*)d_ws;                   // bf16 (q + rel_u), [l*NB+n][e]
  __bf16* kb = qb + QN;                         // bf16 k, [s*NB+n][e]
  __bf16* vt = kb + QN;                         // bf16 v^T, [n][h][d][s]
  float*  op = (float*)(vt + QN);               // fp32 attention out pre-proj
  float*  mb = op + QN;                         // (N*H*L) rowmax
  float*  sb = mb + (size_t)NB * H * L;         // (N*H*L) sumexp

  const int M = L * NB;                          // 8192
  const dim3 blk(256);                           // 8 waves (wave32)
  const dim3 g1(((M / 16) * (E / 16)) / 8);      // GEMM tiles / waves-per-block

  // q/k/v input projections -> bf16 operand caches (rel_u fused into q bias)
  gemm_bt_bias<1><<<g1, blk, 0, stream>>>(query, in_w,             in_b,         rel_u,   qb, M, E, E);
  gemm_bt_bias<1><<<g1, blk, 0, stream>>>(key,   in_w + E * E,     in_b + E,     nullptr, kb, M, E, E);
  gemm_bt_bias<2><<<g1, blk, 0, stream>>>(value, in_w + 2 * E * E, in_b + 2 * E, nullptr, vt, M, E, E);

  // softmax statistics (pass 1)
  const dim3 gs((NB * H * (L / 16)) / 8);        // 512 blocks
  attn_stats<<<gs, blk, 0, stream>>>(qb, kb, mb, sb);

  // probabilities + att@V + head-mean weights (pass 2)
  const dim3 gp(L / 16, NB);
  attn_pass2<<<gp, blk, 0, stream>>>(qb, kb, vt, mb, sb, op, attw);

  // output projection straight into d_out
  gemm_bt_bias<0><<<g1, blk, 0, stream>>>(op, out_w, out_b, nullptr, out, M, E, E);
}
